// output_block1_28647431864533
// MI455X (gfx1250) — compile-verified, matrix operand order checked by reading, not run on hardware
//
#include <hip/hip_runtime.h>

// ---------------------------------------------------------------------------
// Fused pairwise edge-MLP for MI455X (gfx1250, wave32, WMMA).
//   P = X(B*10, 512) @ [W1a|W1b](512,128)    via v_wmma_f32_16x16x32_bf16
//   v[b,p] = W2 . relu(pa[b,i_p] + pb[b,j_p] + b1) + b2
//   out[b] = symmetric 10x10 scatter of v, zero diagonal
// Memory-bound: X read exactly once (1.34 GB -> ~59us @ 23.3 TB/s); bf16 WMMA
// w/ f32 accumulate keeps matrix compute ~2x under the HBM floor.
// This revision: (a) keep global->register prefetch only for the X stream
// (HBM) and drop the B-chunk prefetch registers -- round 2 showed allocation
// past VGPR 255 (s_set_vgpr_msb churn in the WMMA chain, occupancy loss);
// W is L2-resident so its reload needs no register pipeline. (b) NT temporal
// hints on the X stream and the output store so the 1.37 GB stream doesn't
// evict the hot weights from L2/WGP$.
// ---------------------------------------------------------------------------

typedef __attribute__((ext_vector_type(16))) __bf16 v16bf;
typedef __attribute__((ext_vector_type(8)))  __bf16 v8bf;
typedef __attribute__((ext_vector_type(4)))  __bf16 v4bf;
typedef __attribute__((ext_vector_type(2)))  __bf16 v2bf;
typedef __attribute__((ext_vector_type(8)))  float  v8f;
typedef __attribute__((ext_vector_type(4)))  float  v4f;

#define NODES   10
#define DDIM    512
#define HDIM    64
#define NPAIRS  45
#define SPB     16                 // samples per block
#define MROWS   (SPB * NODES)      // 160 rows of X per block
#define KCHUNK  64
#define NCH     (DDIM / KCHUNK)    // 8 K-chunks
#define NCOLS   128                // [W1a|W1b] output width
#define ASTRIDE 72                 // padded bf16 strides (16B-aligned rows)
#define BSTRIDE 72
#define PSTRIDE 132                // padded f32 stride for P spill

// triu_indices(10, k=1), row-major
__device__ __constant__ int c_i[NPAIRS] = {
  0,0,0,0,0,0,0,0,0, 1,1,1,1,1,1,1,1, 2,2,2,2,2,2,2,
  3,3,3,3,3,3, 4,4,4,4,4, 5,5,5,5, 6,6,6, 7,7, 8 };
__device__ __constant__ int c_j[NPAIRS] = {
  1,2,3,4,5,6,7,8,9, 2,3,4,5,6,7,8,9, 3,4,5,6,7,8,9,
  4,5,6,7,8,9, 5,6,7,8,9, 6,7,8,9, 7,8,9, 8,9, 9 };

static __device__ __forceinline__ __bf16 f2bf(float f) {
  unsigned u = __builtin_bit_cast(unsigned, f);
  u += 0x7FFFu + ((u >> 16) & 1u);          // round-to-nearest-even
  unsigned short h = (unsigned short)(u >> 16);
  return __builtin_bit_cast(__bf16, h);
}

__global__ __launch_bounds__(256)
void fused_edge_mlp(const float* __restrict__ X,  const float* __restrict__ W1,
                    const float* __restrict__ B1, const float* __restrict__ W2,
                    const float* __restrict__ B2, float* __restrict__ out, int Bn)
{
  // LDS: staging (A:23040B + B:18432B) aliases P (84480B); tails after P.
  __shared__ __align__(16) unsigned char smem[90112];
  __bf16* Alds = (__bf16*)smem;                        // [160][72] bf16, K contiguous
  __bf16* Blds = (__bf16*)(smem + MROWS * ASTRIDE * 2);// [128][72] bf16, n-major, K contiguous
  float*  Pl   = (float*)smem;                         // [160][132] f32 (after GEMM)
  float*  vbuf = (float*)(smem + 84480);               // [16][45]
  float*  w2s  = (float*)(smem + 87360);               // [64]
  float*  b1s  = (float*)(smem + 87616);               // [64]

  const int tid  = threadIdx.x;
  const int lane = tid & 31;          // wave32
  const int wv   = tid >> 5;          // wave id 0..7 -> column tile
  const int blk  = blockIdx.x;
  const int R0   = blk * MROWS;
  const int totalRows = Bn * NODES;

  if (tid < HDIM) { w2s[tid] = W2[tid]; b1s[tid] = B1[tid]; }

  const v8f vzero = {0.f,0.f,0.f,0.f,0.f,0.f,0.f,0.f};
  v8f acc[10];
  #pragma unroll
  for (int rt = 0; rt < 10; ++rt) acc[rt] = vzero;

  const int rowSel = (lane & 15);       // M (for A) / N (for B) within tile
  const int kOff   = (lane >> 4) << 3;  // lanes 16-31 start at K=8 (ISA layout)

  // ---- X chunk prefetch registers (raw f32; convert only at store time) ----
  v4f xreg[10];

  auto loadX = [&](int c) {
    #pragma unroll
    for (int t = 0; t < 10; ++t) {
      int vid = tid + t * 256;                  // 0..2559
      int m   = vid >> 4;
      int k4  = (vid & 15) << 2;
      int grow = R0 + m;
      if (grow < totalRows)
        xreg[t] = __builtin_nontemporal_load(
            (const v4f*)(X + (size_t)grow * DDIM + c * KCHUNK + k4));
      else
        xreg[t] = (v4f){0.f, 0.f, 0.f, 0.f};
    }
  };
  auto storeX = [&]() {
    #pragma unroll
    for (int t = 0; t < 10; ++t) {
      int vid = tid + t * 256;
      int m   = vid >> 4;
      int k4  = (vid & 15) << 2;
      v4bf bv = { f2bf(xreg[t].x), f2bf(xreg[t].y), f2bf(xreg[t].z), f2bf(xreg[t].w) };
      *(v4bf*)(Alds + m * ASTRIDE + k4) = bv;
    }
  };
  // Wc[k][n] = n<64 ? W1[k][n] : W1[512+k][n-64]; staged transposed (n-major).
  // W is L2-resident (256 KB total) -> plain load+convert+store, no reg pipeline.
  auto stageB = [&](int c) {
    #pragma unroll
    for (int t = 0; t < 16; ++t) {
      int pi = tid + t * 256;                   // 0..4095 -> (n, k-pair)
      int n  = pi & (NCOLS - 1);
      int k2 = (pi >> 7) << 1;
      int gk = c * KCHUNK + k2;
      int row = (n < HDIM) ? gk : (DDIM + gk);
      const float* wp = W1 + (size_t)row * HDIM + (n & (HDIM - 1));
      v2bf bv = { f2bf(wp[0]), f2bf(wp[HDIM]) };
      *(v2bf*)(Blds + n * BSTRIDE + k2) = bv;
    }
  };
  auto loadAfrag = [&](int rt, int kk) -> v16bf {
    const __bf16* aa = Alds + (rt * 16 + rowSel) * ASTRIDE + kk * 32 + kOff;
    v8bf lo = *(const v8bf*)aa;
    v8bf hi = *(const v8bf*)(aa + 16);
    return __builtin_shufflevector(lo, hi, 0,1,2,3,4,5,6,7,8,9,10,11,12,13,14,15);
  };

  // prologue: stage chunk 0
  loadX(0);
  storeX();
  stageB(0);

  #pragma unroll 1
  for (int c = 0; c < NCH; ++c) {
    __syncthreads();                       // staged chunk c visible
    if (c + 1 < NCH) loadX(c + 1);         // HBM stream in flight during WMMA

    #pragma unroll
    for (int kk = 0; kk < 2; ++kk) {
      const __bf16* bb = Blds + (wv * 16 + rowSel) * BSTRIDE + kk * 32 + kOff;
      v8bf blo = *(const v8bf*)bb;
      v8bf bhi = *(const v8bf*)(bb + 16);
      v16bf bf = __builtin_shufflevector(blo, bhi,
                   0,1,2,3,4,5,6,7,8,9,10,11,12,13,14,15);
      v16bf a0 = loadAfrag(0, kk);
      #pragma unroll
      for (int rt = 0; rt < 10; ++rt) {    // pipelined: load rt+1 before wmma rt
        v16bf a1 = (rt < 9) ? loadAfrag(rt + 1, kk) : a0;
        acc[rt] = __builtin_amdgcn_wmma_f32_16x16x32_bf16(
            false, a0, false, bf, (short)0, acc[rt], false, false);
        a0 = a1;
      }
    }
    __syncthreads();                       // all waves done reading chunk c
    if (c + 1 < NCH) { storeX(); stageB(c + 1); }
  }

  // ---- spill accumulators: C/D layout lanes0-15: M=v, lanes16-31: M=v+8 ----
  {
    const int colBase = wv * 16 + rowSel;
    const int rOff    = kOff;              // 0 or 8
    #pragma unroll
    for (int rt = 0; rt < 10; ++rt)
      #pragma unroll
      for (int v = 0; v < 8; ++v)
        Pl[(rt * 16 + rOff + v) * PSTRIDE + colBase] = acc[rt][v];
  }
  __syncthreads();

  // ---- pair MLP head: v = W2 . relu(pa_i + pb_j + b1) + b2 ----
  const float b2v = B2[0];
  for (int q = tid; q < SPB * NPAIRS; q += 256) {
    int s = q / NPAIRS, p = q - s * NPAIRS;
    if (blk * SPB + s < Bn) {
      const float* pa = Pl + (s * NODES + c_i[p]) * PSTRIDE;
      const float* pb = Pl + (s * NODES + c_j[p]) * PSTRIDE + HDIM;
      float a = b2v;
      #pragma unroll
      for (int h = 0; h < HDIM; ++h) {
        float t = pa[h] + pb[h] + b1s[h];
        a = fmaf(fmaxf(t, 0.f), w2s[h], a);
      }
      vbuf[q] = a;
    }
  }
  __syncthreads();

  // ---- symmetric scatter to out[B,10,10], zero diagonal (NT stores) ----
  for (int idx = tid; idx < SPB * NODES * NODES; idx += 256) {
    int s  = idx / (NODES * NODES);
    int cc = idx - s * NODES * NODES;
    int gs = blk * SPB + s;
    if (gs >= Bn) break;                   // gs nondecreasing with idx
    int i = cc / NODES, j = cc - i * NODES;
    float val = 0.f;
    if (i != j) {
      int a = i < j ? i : j, b = i < j ? j : i;
      int p = a * (2 * NODES - 1 - a) / 2 + (b - a - 1);
      val = vbuf[s * NPAIRS + p];
    }
    __builtin_nontemporal_store(val, out + (size_t)gs * NODES * NODES + cc);
  }
}

extern "C" void kernel_launch(void* const* d_in, const int* in_sizes, int n_in,
                              void* d_out, int out_size, void* d_ws, size_t ws_size,
                              hipStream_t stream) {
  const float* X  = (const float*)d_in[0];
  const float* W1 = (const float*)d_in[1];
  const float* B1 = (const float*)d_in[2];
  const float* W2 = (const float*)d_in[3];
  const float* B2 = (const float*)d_in[4];
  float* out = (float*)d_out;
  (void)d_ws; (void)ws_size; (void)n_in; (void)out_size;

  int Bn = in_sizes[0] / (NODES * DDIM);           // 65536
  int blocks = (Bn + SPB - 1) / SPB;               // 4096
  hipLaunchKernelGGL(fused_edge_mlp, dim3(blocks), dim3(256), 0, stream,
                     X, W1, B1, W2, B2, out, Bn);
}
